// Attention_53652731461991
// MI455X (gfx1250) — matrix operand order repaired and checked by reading, not compiled, not run on hardware
//
#include <hip/hip_runtime.h>

// ---------------------------------------------------------------------------
// Fused attention block for MI455X (gfx1250, wave32, WMMA 16x16x32 bf16)
//   qkv = x @ w_qkv + b_qkv ; LN(q)*hd^-0.5, LN(k) ; softmax(q k^T) v ; @ w_proj
// ---------------------------------------------------------------------------

typedef __bf16 bf16;
typedef __attribute__((ext_vector_type(16))) __bf16 v16bf;
typedef __attribute__((ext_vector_type(8)))  __bf16 v8bf;
typedef __attribute__((ext_vector_type(4)))  __bf16 v4bf;
typedef __attribute__((ext_vector_type(2)))  __bf16 v2bf;
typedef __attribute__((ext_vector_type(8)))  float  v8f;

#define WMMA_BF16(a, b, c) \
  __builtin_amdgcn_wmma_f32_16x16x32_bf16(false, (a), false, (b), (short)0, (c), false, false)

// ---- fragment loaders -----------------------------------------------------
// A fragment 16x32 (bf16), row-major source, lda in elements.
// Lane l: row m = l&15, half = l>>4. VGPR0-3: K = (0..7)+half*8 ; VGPR4-7: K = 16+(0..7)+half*8
__device__ __forceinline__ v16bf load_frag_a(const bf16* __restrict__ base, int lda,
                                             int k0, int lane) {
  const int half = lane >> 4, m = lane & 15;
  const bf16* p = base + (size_t)m * lda + k0 + half * 8;
  v8bf lo = *(const v8bf*)p;         // K = k0 + half*8 + [0..7]
  v8bf hi = *(const v8bf*)(p + 16);  // K = k0 + 16 + half*8 + [0..7]
  v16bf r;
#pragma unroll
  for (int i = 0; i < 8; ++i) { r[i] = lo[i]; r[i + 8] = hi[i]; }
  return r;
}

// B fragment 32x16 (bf16) from an N-major (transposed) operand BT[n][k], ldb in elements.
// Lane l: column n = n0 + (l&15); K elements = k0 + (l>>4)*8 + [0..7], k0 + 16 + (l>>4)*8 + [0..7].
__device__ __forceinline__ v16bf load_frag_b(const bf16* __restrict__ baseT, int ldb,
                                             int n0, int k0, int lane) {
  const int half = lane >> 4, n = lane & 15;
  const bf16* p = baseT + (size_t)(n0 + n) * ldb + k0 + half * 8;
  v8bf lo = *(const v8bf*)p;
  v8bf hi = *(const v8bf*)(p + 16);
  v16bf r;
#pragma unroll
  for (int i = 0; i < 8; ++i) { r[i] = lo[i]; r[i + 8] = hi[i]; }
  return r;
}

// ---- wave reductions (wave32) ---------------------------------------------
__device__ __forceinline__ float red_max16(float v) {   // across each 16-lane half
#pragma unroll
  for (int m = 1; m < 16; m <<= 1) v = fmaxf(v, __shfl_xor(v, m, 32));
  return v;
}
__device__ __forceinline__ float red_sum16(float v) {
#pragma unroll
  for (int m = 1; m < 16; m <<= 1) v += __shfl_xor(v, m, 32);
  return v;
}
__device__ __forceinline__ float red_sum32(float v) {
#pragma unroll
  for (int m = 16; m >= 1; m >>= 1) v += __shfl_xor(v, m, 32);
  return v;
}

// ---------------------------------------------------------------------------
// 1) fp32 -> bf16 pack (vectorized)
// ---------------------------------------------------------------------------
__global__ __launch_bounds__(256) void pack_bf16_kernel(const float* __restrict__ in,
                                                        bf16* __restrict__ out, int n4) {
  int i = blockIdx.x * 256 + threadIdx.x;
  if (i < n4) {
    float4 x = ((const float4*)in)[i];
    v4bf o;
    o[0] = (bf16)x.x; o[1] = (bf16)x.y; o[2] = (bf16)x.z; o[3] = (bf16)x.w;
    ((v4bf*)out)[i] = o;
  }
}

// ---------------------------------------------------------------------------
// 2) Transpose-pack weights: w[K][N] fp32 -> wT[N][K] bf16  (32x32 LDS tiles)
// ---------------------------------------------------------------------------
__global__ __launch_bounds__(256) void transpose_pack_kernel(const float* __restrict__ w,
                                                             bf16* __restrict__ wT,
                                                             int K, int N) {
  __shared__ __bf16 tile[32][33];
  const int kb = blockIdx.x * 32, nb = blockIdx.y * 32;
  const int tx = threadIdx.x & 31, ty = threadIdx.x >> 5;   // 8 rows per pass
#pragma unroll
  for (int r = ty; r < 32; r += 8)
    tile[r][tx] = (bf16)w[(size_t)(kb + r) * N + nb + tx];
  __syncthreads();
#pragma unroll
  for (int r = ty; r < 32; r += 8)
    wT[(size_t)(nb + r) * K + kb + tx] = tile[tx][r];
}

// ---------------------------------------------------------------------------
// 3) GEMM: C[M][N] (fp32, +bias) = A[M][K](bf16, row-major) * BT[N][K](bf16)
//    Block = 8 waves (4 along M x 2 along N) -> 128x128 tile.
//    Wave -> 32x64 (2x4 WMMA tiles, 8 WMMA / K-step) => ~21 FLOP/byte of
//    fragment traffic, keeping the K-loop matrix-core bound rather than
//    L2-bound, while staying under the 256-VGPR direct-address budget.
// ---------------------------------------------------------------------------
__global__ __launch_bounds__(256) void gemm_bf16_kernel(const bf16* __restrict__ A,
                                                        const bf16* __restrict__ BT,
                                                        const float* __restrict__ bias,
                                                        float* __restrict__ C,
                                                        int M, int N, int K) {
  const int lane = threadIdx.x & 31;
  const int wave = threadIdx.x >> 5;
  const int half = lane >> 4, l15 = lane & 15;
  const int m0 = blockIdx.x * 128 + (wave & 3) * 32;
  const int n0 = blockIdx.y * 128 + (wave >> 2) * 64;

  const bf16* A0 = A + (size_t)m0 * K;
  const bf16* A1 = A0 + (size_t)16 * K;

  v8f acc0[4] = {}, acc1[4] = {};
  for (int k0 = 0; k0 < K; k0 += 32) {
    v16bf a0 = load_frag_a(A0, K, k0, lane);
    v16bf a1 = load_frag_a(A1, K, k0, lane);
#pragma unroll
    for (int j = 0; j < 4; ++j) {
      v16bf bj = load_frag_b(BT, K, n0 + 16 * j, k0, lane);
      acc0[j] = WMMA_BF16(a0, bj, acc0[j]);
      acc1[j] = WMMA_BF16(a1, bj, acc1[j]);
    }
  }
#pragma unroll
  for (int j = 0; j < 4; ++j) {
    const float bia = bias[n0 + 16 * j + l15];
#pragma unroll
    for (int r = 0; r < 8; ++r) {
      const int row0 = m0 + r + 8 * half;
      C[(size_t)row0 * N + n0 + 16 * j + l15]        = acc0[j][r] + bia;
      C[(size_t)(row0 + 16) * N + n0 + 16 * j + l15] = acc1[j][r] + bia;
    }
  }
}

// ---------------------------------------------------------------------------
// 4) LayerNorm(q)*hd^-0.5, LayerNorm(k), split heads, pack bf16.
//    One wave per (b,h,n) row of 64; 2 elements per lane; shuffle reductions.
//    qhat/khat: [B,H,N,64] row-major.  vT: [B,H,64,N] (key-contiguous for PV).
// ---------------------------------------------------------------------------
__global__ __launch_bounds__(256) void ln_pack_kernel(const float* __restrict__ qkv,
                                                      const float* __restrict__ gq,
                                                      const float* __restrict__ bq,
                                                      const float* __restrict__ gk,
                                                      const float* __restrict__ bk,
                                                      bf16* __restrict__ qhat,
                                                      bf16* __restrict__ khat,
                                                      bf16* __restrict__ vT) {
  const int lane = threadIdx.x & 31;
  const int wave = threadIdx.x >> 5;
  const int t = blockIdx.x * 8 + wave;          // row id: (b*16+h)*2048 + n
  const int n = t & 2047;
  const int bh = t >> 11;                       // b*16 + h
  const int b = t >> 15, h = (t >> 11) & 15;

  const float* q = qkv + ((size_t)(b * 2048 + n)) * 3072 + h * 64;
  const float* k = q + 1024;
  const float* v = q + 2048;
  const int d0 = lane * 2;

  float2 qv = *(const float2*)(q + d0);
  float2 kv = *(const float2*)(k + d0);
  float2 vv = *(const float2*)(v + d0);

  float qs = red_sum32(qv.x + qv.y);
  float qss = red_sum32(qv.x * qv.x + qv.y * qv.y);
  float ks = red_sum32(kv.x + kv.y);
  float kss = red_sum32(kv.x * kv.x + kv.y * kv.y);

  const float inv64 = 1.0f / 64.0f;
  float qmu = qs * inv64, kmu = ks * inv64;
  float qr = rsqrtf(qss * inv64 - qmu * qmu + 1e-5f);
  float kr = rsqrtf(kss * inv64 - kmu * kmu + 1e-5f);
  const float qscale = 0.125f;                  // 64^-0.5

  float2 g2q = *(const float2*)(gq + d0), b2q = *(const float2*)(bq + d0);
  float2 g2k = *(const float2*)(gk + d0), b2k = *(const float2*)(bk + d0);

  v2bf qo, ko;
  qo[0] = (bf16)(((qv.x - qmu) * qr * g2q.x + b2q.x) * qscale);
  qo[1] = (bf16)(((qv.y - qmu) * qr * g2q.y + b2q.y) * qscale);
  ko[0] = (bf16)((kv.x - kmu) * kr * g2k.x + b2k.x);
  ko[1] = (bf16)((kv.y - kmu) * kr * g2k.y + b2k.y);
  *(v2bf*)(qhat + (size_t)t * 64 + d0) = qo;
  *(v2bf*)(khat + (size_t)t * 64 + d0) = ko;

  bf16* vbase = vT + (size_t)bh * 64 * 2048 + n;
  vbase[(size_t)d0 * 2048]       = (bf16)vv.x;
  vbase[(size_t)(d0 + 1) * 2048] = (bf16)vv.y;
}

// ---------------------------------------------------------------------------
// 5) Flash attention: one wave per 16-query tile; stream 32-key chunks.
//    S = Q K^T (2 WMMA per 16x16 S-tile, hd=64) ; online softmax ;
//    P (D-layout) -> LDS -> A-layout ; O += P V (4 WMMA per chunk).
// ---------------------------------------------------------------------------
__global__ __launch_bounds__(256) void attn_kernel(const bf16* __restrict__ qhat,
                                                   const bf16* __restrict__ khat,
                                                   const bf16* __restrict__ vT,
                                                   bf16* __restrict__ out) {
  __shared__ __bf16 plds_all[8 * 16 * 32];      // 1KB per wave
  const int lane = threadIdx.x & 31;
  const int wave = threadIdx.x >> 5;
  __bf16* plds = plds_all + wave * 512;

  const int w = blockIdx.x * 8 + wave;          // 0 .. B*H*(N/16)-1
  const int qt = w & 127;                       // query tile (N/16 = 128)
  const int bh = w >> 7;                        // b*16 + h
  const int N = 2048, HD = 64;
  const int half = lane >> 4, l15 = lane & 15;

  const bf16* qbase = qhat + ((size_t)bh * N + qt * 16) * HD;
  const bf16* kbase = khat + (size_t)bh * N * HD;
  const bf16* vbase = vT + (size_t)bh * HD * N;

  // resident Q fragments (16 x 64)
  v16bf aq0 = load_frag_a(qbase, HD, 0, lane);
  v16bf aq1 = load_frag_a(qbase, HD, 32, lane);

  v8f o0 = {}, o1 = {}, o2 = {}, o3 = {};
  float mrow[8], lrow[8];
#pragma unroll
  for (int r = 0; r < 8; ++r) { mrow[r] = -1e30f; lrow[r] = 0.0f; }

  for (int key0 = 0; key0 < N; key0 += 32) {
    v8f S0 = {}, S1 = {};
    S0 = WMMA_BF16(aq0, load_frag_b(kbase, HD, key0, 0, lane), S0);
    S0 = WMMA_BF16(aq1, load_frag_b(kbase, HD, key0, 32, lane), S0);
    S1 = WMMA_BF16(aq0, load_frag_b(kbase, HD, key0 + 16, 0, lane), S1);
    S1 = WMMA_BF16(aq1, load_frag_b(kbase, HD, key0 + 16, 32, lane), S1);

#pragma unroll
    for (int r = 0; r < 8; ++r) {
      float smax = red_max16(fmaxf(S0[r], S1[r]));
      float mnew = fmaxf(mrow[r], smax);
      float corr = __expf(mrow[r] - mnew);
      mrow[r] = mnew;
      float p0 = __expf(S0[r] - mnew);
      float p1 = __expf(S1[r] - mnew);
      lrow[r] = lrow[r] * corr + red_sum16(p0 + p1);
      o0[r] *= corr; o1[r] *= corr; o2[r] *= corr; o3[r] *= corr;
      const int row = r + 8 * half;
      plds[row * 32 + l15]      = (bf16)p0;     // D-layout -> row-major 16x32 tile
      plds[row * 32 + 16 + l15] = (bf16)p1;
    }
    asm volatile("s_wait_dscnt 0" ::: "memory"); // LDS stores visible before reload

    // reload P in A-fragment layout (same-wave, LDS in-order)
    v16bf pf;
    {
      const __bf16* p = plds + l15 * 32 + half * 8;
      v8bf lo = *(const v8bf*)p;
      v8bf hi = *(const v8bf*)(p + 16);
#pragma unroll
      for (int i = 0; i < 8; ++i) { pf[i] = lo[i]; pf[i + 8] = hi[i]; }
    }
    asm volatile("s_wait_dscnt 0" ::: "memory");

    o0 = WMMA_BF16(pf, load_frag_b(vbase, N, 0, key0, lane), o0);
    o1 = WMMA_BF16(pf, load_frag_b(vbase, N, 16, key0, lane), o1);
    o2 = WMMA_BF16(pf, load_frag_b(vbase, N, 32, key0, lane), o2);
    o3 = WMMA_BF16(pf, load_frag_b(vbase, N, 48, key0, lane), o3);
  }

  // epilogue: normalize and scatter to token-major [B*N, C] bf16 for proj GEMM
  const int b = bh >> 4, h = bh & 15;
#pragma unroll
  for (int r = 0; r < 8; ++r) {
    const float inv = 1.0f / lrow[r];
    const int tok = b * N + qt * 16 + r + 8 * half;
    bf16* orow = out + (size_t)tok * 1024 + h * 64;
    orow[l15]      = (bf16)(o0[r] * inv);
    orow[16 + l15] = (bf16)(o1[r] * inv);
    orow[32 + l15] = (bf16)(o2[r] * inv);
    orow[48 + l15] = (bf16)(o3[r] * inv);
  }
}

// ---------------------------------------------------------------------------
extern "C" void kernel_launch(void* const* d_in, const int* in_sizes, int n_in,
                              void* d_out, int out_size, void* d_ws, size_t ws_size,
                              hipStream_t stream) {
  (void)in_sizes; (void)n_in; (void)out_size; (void)ws_size;
  const float* x      = (const float*)d_in[0];
  const float* w_qkv  = (const float*)d_in[1];
  const float* b_qkv  = (const float*)d_in[2];
  const float* g_q    = (const float*)d_in[3];
  const float* be_q   = (const float*)d_in[4];
  const float* g_k    = (const float*)d_in[5];
  const float* be_k   = (const float*)d_in[6];
  const float* w_proj = (const float*)d_in[7];
  const float* b_proj = (const float*)d_in[8];
  float* out = (float*)d_out;

  const int Btok = 4096;             // B*N
  const int C = 1024, C3 = 3072;

  char* ws = (char*)d_ws;
  size_t off = 0;
  auto alloc = [&](size_t bytes) -> char* {
    char* p = ws + off;
    off = (off + bytes + 255) & ~(size_t)255;
    return p;
  };
  bf16*  xb     = (bf16*)alloc((size_t)Btok * C * 2);   // x in bf16
  bf16*  wqkvT  = (bf16*)alloc((size_t)C3 * C * 2);     // w_qkv^T bf16
  bf16*  wprojT = (bf16*)alloc((size_t)C * C * 2);      // w_proj^T bf16
  float* qkv    = (float*)alloc((size_t)Btok * C3 * 4); // fp32 qkv
  bf16*  qhat   = (bf16*)alloc((size_t)Btok * C * 2);   // [B,H,N,64]
  bf16*  khat   = (bf16*)alloc((size_t)Btok * C * 2);   // [B,H,N,64]
  bf16*  vTbuf  = (bf16*)alloc((size_t)Btok * C * 2);   // [B,H,64,N]
  bf16*  aout   = (bf16*)alloc((size_t)Btok * C * 2);   // attn out, token-major

  pack_bf16_kernel<<<(Btok * C / 4) / 256, 256, 0, stream>>>(x, xb, Btok * C / 4);
  transpose_pack_kernel<<<dim3(C / 32, C3 / 32), 256, 0, stream>>>(w_qkv, wqkvT, C, C3);
  transpose_pack_kernel<<<dim3(C / 32, C / 32), 256, 0, stream>>>(w_proj, wprojT, C, C);

  // QKV GEMM: M=4096, N=3072, K=1024 ; block tile 128x128
  gemm_bf16_kernel<<<dim3(Btok / 128, C3 / 128), 256, 0, stream>>>(
      xb, wqkvT, b_qkv, qkv, Btok, C3, C);

  // 65536 (b,h,n) rows, 8 waves/block -> 8192 blocks
  ln_pack_kernel<<<8192, 256, 0, stream>>>(
      qkv, g_q, be_q, g_k, be_k, qhat, khat, vTbuf);

  attn_kernel<<<(2 * 16 * 128) / 8, 256, 0, stream>>>(qhat, khat, vTbuf, aout);

  // Proj GEMM: M=4096, N=1024, K=1024
  gemm_bf16_kernel<<<dim3(Btok / 128, C / 128), 256, 0, stream>>>(
      aout, wprojT, b_proj, out, Btok, C, C);
}
